// Attention_72395968741716
// MI455X (gfx1250) — matrix operand order repaired and checked by reading, not compiled, not run on hardware
//
#include <hip/hip_runtime.h>

typedef _Float16 half_t;
typedef __attribute__((ext_vector_type(16))) _Float16 v16h;
typedef __attribute__((ext_vector_type(8)))  _Float16 v8h;
typedef __attribute__((ext_vector_type(8)))  float    v8f;

#define B_     4
#define T_     2048
#define D_     1024
#define H_     16
#define DH_    64
#define INNER_ 1024
#define ROWS_  (B_ * T_)
#define SCALE_ 0.125f   /* DH^-0.5 */

// ---------------------------------------------------------------------------
// WMMA helpers (CDNA5 16x16x32 f16 -> f32)
// ---------------------------------------------------------------------------
__device__ __forceinline__ v8f wmma_f16(v16h a, v16h b, v8f c) {
  return __builtin_amdgcn_wmma_f32_16x16x32_f16(false, a, false, b, (short)0, c,
                                                false, false);
}

// A fragment: 16x32 (MxK), row-major source, leading dim `ld` (halfs).
// lanes 0-15 -> M=lane, halves 0..15 = K 0..15;
// lanes 16-31 -> same M, K 16..31.
__device__ __forceinline__ v16h load_a(const half_t* base, int ld, int lane) {
  int m  = lane & 15;
  int hi = (lane >> 4) & 1;
  const half_t* p = base + (size_t)m * ld + hi * 16;
  return *(const v16h*)(p);
}

// B fragment: 32x16 (KxN) from N-major ("transposed") storage, leading dim `ld`.
// Lane n = N column; halves h -> K = h + 16*(lane>=16). One 32B load per lane.
__device__ __forceinline__ v16h load_bt(const half_t* baseT, int ld, int lane) {
  int n  = lane & 15;
  int hi = (lane >> 4) & 1;
  return *(const v16h*)(baseT + (size_t)n * ld + hi * 16);
}

// CDNA5 async copy: 16B global -> LDS, tracked by ASYNCcnt.
__device__ __forceinline__ void async_ld16(unsigned lds_off, const half_t* g) {
  asm volatile("global_load_async_to_lds_b128 %0, %1, off"
               :: "v"(lds_off), "v"(g) : "memory");
}

// ---------------------------------------------------------------------------
// Precision conversion kernels
// ---------------------------------------------------------------------------
__global__ void cvt_f16_kernel(const float* __restrict__ src, half_t* __restrict__ dst, int n) {
  int i      = blockIdx.x * blockDim.x + threadIdx.x;
  int stride = gridDim.x * blockDim.x;
  for (; i < n; i += stride) dst[i] = (half_t)src[i];
}

// src: K x N (f32, row-major) -> dst: N x K (f16, row-major)
__global__ void transpose_f16_kernel(const float* __restrict__ src, half_t* __restrict__ dst,
                                     int K, int N) {
  int i      = blockIdx.x * blockDim.x + threadIdx.x;
  int stride = gridDim.x * blockDim.x;
  int total  = K * N;
  for (; i < total; i += stride) {
    int nn = i / K;
    int kk = i - nn * K;
    dst[i] = (half_t)src[(size_t)kk * N + nn];
  }
}

// ---------------------------------------------------------------------------
// Fused QKV projection, 32 rows x 16 cols per wave (2 A-frags amortize 3 B-frags)
// Outputs Q,K as [B,H,T,DH] and V transposed [B,H,DH,T].
// ---------------------------------------------------------------------------
__global__ void qkv_kernel(const half_t* __restrict__ xh,
                           const half_t* __restrict__ wqT,
                           const half_t* __restrict__ wkT,
                           const half_t* __restrict__ wvT,
                           half_t* __restrict__ Q,
                           half_t* __restrict__ Kd,
                           half_t* __restrict__ VT) {
  int lane  = threadIdx.x & 31;
  int wave  = threadIdx.x >> 5;
  int tile  = blockIdx.x * 4 + wave;           // (ROWS/32)*(INNER/16) tiles
  int ntile = tile & (INNER_ / 16 - 1);
  int mt2   = tile >> 6;                       // 32-row supertile

  const half_t* arow = xh  + (size_t)mt2 * 32 * D_;
  const half_t* bq   = wqT + (size_t)ntile * 16 * D_;
  const half_t* bk   = wkT + (size_t)ntile * 16 * D_;
  const half_t* bv   = wvT + (size_t)ntile * 16 * D_;

  v8f cq0 = {}, ck0 = {}, cv0 = {}, cq1 = {}, ck1 = {}, cv1 = {};
  for (int ks = 0; ks < D_; ks += 32) {
    if (ks + 64 < D_) {
      __builtin_prefetch(arow + ks + 64, 0, 1);
      __builtin_prefetch(bq + ks + 64, 0, 1);
    }
    v16h a0 = load_a(arow + ks, D_, lane);
    v16h a1 = load_a(arow + 16 * D_ + ks, D_, lane);
    v16h fq = load_bt(bq + ks, D_, lane);
    v16h fk = load_bt(bk + ks, D_, lane);
    v16h fv = load_bt(bv + ks, D_, lane);
    cq0 = wmma_f16(a0, fq, cq0);  cq1 = wmma_f16(a1, fq, cq1);
    ck0 = wmma_f16(a0, fk, ck0);  ck1 = wmma_f16(a1, fk, ck1);
    cv0 = wmma_f16(a0, fv, cv0);  cv1 = wmma_f16(a1, fv, cv1);
  }

  int n  = lane & 15;
  int hi = (lane >> 4) & 1;
  int col = ntile * 16 + n;
  int h   = col >> 6;
  int d   = col & (DH_ - 1);
#pragma unroll
  for (int half = 0; half < 2; ++half) {
#pragma unroll
    for (int r = 0; r < 8; ++r) {
      int row = mt2 * 32 + half * 16 + r + 8 * hi;
      int b   = row >> 11;
      int t   = row & (T_ - 1);
      size_t qi = (((size_t)(b * H_ + h) * T_) + t) * DH_ + d;
      float vq = half ? cq1[r] : cq0[r];
      float vk = half ? ck1[r] : ck0[r];
      float vv = half ? cv1[r] : cv0[r];
      Q[qi]  = (half_t)vq;
      Kd[qi] = (half_t)vk;
      VT[((size_t)(b * H_ + h) * DH_ + d) * T_ + t] = (half_t)vv;
    }
  }
}

// ---------------------------------------------------------------------------
// Flash attention. Block = 4 waves working on 4 q-tiles of the SAME (b,h).
// K/V chunks (32 keys) are double-buffered in LDS via async-to-LDS copies
// (ASYNCcnt), shared by all 4 waves -> 4x less L2 traffic.
// ---------------------------------------------------------------------------
__global__ void attn_kernel(const half_t* __restrict__ Q,
                            const half_t* __restrict__ Kd,
                            const half_t* __restrict__ VT,
                            const float*  __restrict__ mask,
                            half_t* __restrict__ attnH) {
  __shared__ __align__(16) half_t ldsK[2][32 * DH_];   // [key][d]   4KB each
  __shared__ __align__(16) half_t ldsV[2][DH_ * 32];   // [d][key]   4KB each
  __shared__ __align__(16) half_t ldsP[4][16 * 32];    // per-wave P

  int lane = threadIdx.x & 31;
  int wave = threadIdx.x >> 5;
  int t    = threadIdx.x;

  int blk = blockIdx.x;                 // B*H*(T/64) blocks
  int qg  = blk & (T_ / 64 - 1);
  int bh  = blk >> 5;                   // T_/64 == 32
  int b   = bh >> 4;
  int hh  = bh & (H_ - 1);
  int qt  = qg * 4 + wave;

  const half_t* Qb = Q  + ((size_t)bh * T_ + (size_t)qt * 16) * DH_;
  const half_t* Kb = Kd + (size_t)bh * T_ * DH_;
  const half_t* Vb = VT + (size_t)bh * DH_ * T_;
  const float*  Mb = mask + ((size_t)b * T_ + (size_t)qt * 16) * T_;

  int n  = lane & 15;
  int hi = (lane >> 4) & 1;

  v16h aq0 = load_a(Qb + 0,  DH_, lane);
  v16h aq1 = load_a(Qb + 32, DH_, lane);

  v8f o0 = {}, o1 = {}, o2 = {}, o3 = {};
  float mrow[8], lrow[8];
#pragma unroll
  for (int r = 0; r < 8; ++r) { mrow[r] = -1e30f; lrow[r] = 0.f; }

  half_t* P = ldsP[wave];

  // cooperative staging: thread copies 32B of K tile and 32B of V tile
  int vr = t >> 1;                 // V^T row (d), 0..63
  int vh = (t & 1) * 16;           // 16-half slice within row
  auto stage = [&](int kc, int buf) {
    const half_t* gk = Kb + (size_t)kc * DH_ + t * 16;
    unsigned ko = (unsigned)(uintptr_t)(&ldsK[buf][t * 16]);
    async_ld16(ko,      gk);
    async_ld16(ko + 16, gk + 8);
    const half_t* gv = Vb + (size_t)vr * T_ + kc + vh;
    unsigned vo = (unsigned)(uintptr_t)(&ldsV[buf][vr * 32 + vh]);
    async_ld16(vo,      gv);
    async_ld16(vo + 16, gv + 8);
  };

  const int NCH = T_ / 32;
  stage(0, 0);
  int cur = 0;
  for (int i = 0; i < NCH; ++i) {
    int kc = i * 32;
    if (i + 1 < NCH) {
      stage((i + 1) * 32, cur ^ 1);
      asm volatile("s_wait_asynccnt 4" ::: "memory");   // chunk i landed
    } else {
      asm volatile("s_wait_asynccnt 0" ::: "memory");
    }
    __syncthreads();                                    // all waves' writes visible

    const half_t* kt = ldsK[cur];
    const half_t* vt = ldsV[cur];

    // ---- S = Q * K^T for 32 keys (two 16x16 C tiles) ----
    v8f s0 = {}, s1 = {};
    s0 = wmma_f16(aq0, load_bt(kt + 0 * 16 * DH_ + 0,  DH_, lane), s0);
    s0 = wmma_f16(aq1, load_bt(kt + 0 * 16 * DH_ + 32, DH_, lane), s0);
    s1 = wmma_f16(aq0, load_bt(kt + 1 * 16 * DH_ + 0,  DH_, lane), s1);
    s1 = wmma_f16(aq1, load_bt(kt + 1 * 16 * DH_ + 32, DH_, lane), s1);

    // ---- online softmax over this 32-key chunk ----
#pragma unroll
    for (int r = 0; r < 8; ++r) {
      int m = r + 8 * hi;
      float p0 = s0[r] * SCALE_ + Mb[(size_t)m * T_ + kc + n];
      float p1 = s1[r] * SCALE_ + Mb[(size_t)m * T_ + kc + 16 + n];
      float loc = fmaxf(p0, p1);
#pragma unroll
      for (int off = 1; off < 16; off <<= 1)
        loc = fmaxf(loc, __shfl_xor(loc, off, 32));
      float newm = fmaxf(mrow[r], loc);
      float corr = __expf(mrow[r] - newm);
      mrow[r] = newm;
      p0 = __expf(p0 - newm);
      p1 = __expf(p1 - newm);
      float rs = p0 + p1;
#pragma unroll
      for (int off = 1; off < 16; off <<= 1)
        rs += __shfl_xor(rs, off, 32);
      lrow[r] = lrow[r] * corr + rs;
      o0[r] *= corr; o1[r] *= corr; o2[r] *= corr; o3[r] *= corr;
      P[m * 32 + n]      = (half_t)p0;
      P[m * 32 + 16 + n] = (half_t)p1;
    }

    // same-wave LDS RAW on P: DS is in-order; block compile-time reordering
    __builtin_amdgcn_wave_barrier();
    asm volatile("s_wait_dscnt 0" ::: "memory");
    v16h ap = load_a(P, 32, lane);
    __builtin_amdgcn_wave_barrier();

    // ---- O += P * V (4 d-tiles of 16) ----
    o0 = wmma_f16(ap, load_bt(vt + 0 * 16 * 32, 32, lane), o0);
    o1 = wmma_f16(ap, load_bt(vt + 1 * 16 * 32, 32, lane), o1);
    o2 = wmma_f16(ap, load_bt(vt + 2 * 16 * 32, 32, lane), o2);
    o3 = wmma_f16(ap, load_bt(vt + 3 * 16 * 32, 32, lane), o3);

    __syncthreads();      // everyone done reading buf[cur] before it is refilled
    cur ^= 1;
  }

  // ---- normalize + store as [B,T,H,DH] ----
#pragma unroll
  for (int r = 0; r < 8; ++r) {
    int m = r + 8 * hi;
    float inv = 1.f / lrow[r];
    int tt = qt * 16 + m;
    size_t base = (((size_t)b * T_ + tt) * H_ + hh) * DH_;
    attnH[base + 0 * 16 + n] = (half_t)(o0[r] * inv);
    attnH[base + 1 * 16 + n] = (half_t)(o1[r] * inv);
    attnH[base + 2 * 16 + n] = (half_t)(o2[r] * inv);
    attnH[base + 3 * 16 + n] = (half_t)(o3[r] * inv);
  }
}

// ---------------------------------------------------------------------------
// Output projection, 32x32 tile per wave: [ROWS, INNER] x [INNER, D] + bo
// ---------------------------------------------------------------------------
__global__ void out_proj_kernel(const half_t* __restrict__ attnH,
                                const half_t* __restrict__ woT,
                                const float*  __restrict__ bo,
                                float* __restrict__ out) {
  int lane  = threadIdx.x & 31;
  int wave  = threadIdx.x >> 5;
  int tile  = blockIdx.x * 4 + wave;           // (ROWS/32)*(D/32) tiles
  int nt2   = tile & (D_ / 32 - 1);
  int mt2   = tile >> 5;

  const half_t* arow = attnH + (size_t)mt2 * 32 * INNER_;
  const half_t* bw   = woT   + (size_t)nt2 * 32 * INNER_;

  v8f c00 = {}, c01 = {}, c10 = {}, c11 = {};
  for (int ks = 0; ks < INNER_; ks += 32) {
    if (ks + 64 < INNER_) {
      __builtin_prefetch(arow + ks + 64, 0, 1);
      __builtin_prefetch(bw + ks + 64, 0, 1);
    }
    v16h a0 = load_a(arow + ks, INNER_, lane);
    v16h a1 = load_a(arow + 16 * INNER_ + ks, INNER_, lane);
    v16h b0 = load_bt(bw + ks, INNER_, lane);
    v16h b1 = load_bt(bw + 16 * INNER_ + ks, INNER_, lane);
    c00 = wmma_f16(a0, b0, c00);
    c01 = wmma_f16(a0, b1, c01);
    c10 = wmma_f16(a1, b0, c10);
    c11 = wmma_f16(a1, b1, c11);
  }

  int n  = lane & 15;
  int hi = (lane >> 4) & 1;
  float bias0 = bo[nt2 * 32 + n];
  float bias1 = bo[nt2 * 32 + 16 + n];
#pragma unroll
  for (int r = 0; r < 8; ++r) {
    int row0 = mt2 * 32 + r + 8 * hi;
    int row1 = row0 + 16;
    out[(size_t)row0 * D_ + nt2 * 32 + n]      = c00[r] + bias0;
    out[(size_t)row0 * D_ + nt2 * 32 + 16 + n] = c01[r] + bias1;
    out[(size_t)row1 * D_ + nt2 * 32 + n]      = c10[r] + bias0;
    out[(size_t)row1 * D_ + nt2 * 32 + 16 + n] = c11[r] + bias1;
  }
}

// ---------------------------------------------------------------------------
// Launch
// ---------------------------------------------------------------------------
extern "C" void kernel_launch(void* const* d_in, const int* in_sizes, int n_in,
                              void* d_out, int out_size, void* d_ws, size_t ws_size,
                              hipStream_t stream) {
  (void)in_sizes; (void)n_in; (void)out_size; (void)ws_size;
  const float* x    = (const float*)d_in[0];
  const float* mask = (const float*)d_in[1];
  const float* Wq   = (const float*)d_in[2];
  const float* Wk   = (const float*)d_in[3];
  const float* Wv   = (const float*)d_in[4];
  const float* Wo   = (const float*)d_in[5];
  const float* bo   = (const float*)d_in[6];
  float* out = (float*)d_out;

  char*  ws  = (char*)d_ws;
  size_t off = 0;
  auto alloc = [&](size_t bytes) -> void* {
    void* p = ws + off;
    off = (off + bytes + 255) & ~(size_t)255;
    return p;
  };
  half_t* xh    = (half_t*)alloc((size_t)ROWS_ * D_ * sizeof(half_t));
  half_t* wqT   = (half_t*)alloc((size_t)D_ * INNER_ * sizeof(half_t));
  half_t* wkT   = (half_t*)alloc((size_t)D_ * INNER_ * sizeof(half_t));
  half_t* wvT   = (half_t*)alloc((size_t)D_ * INNER_ * sizeof(half_t));
  half_t* woT   = (half_t*)alloc((size_t)INNER_ * D_ * sizeof(half_t));
  half_t* Q     = (half_t*)alloc((size_t)ROWS_ * INNER_ * sizeof(half_t));
  half_t* K     = (half_t*)alloc((size_t)ROWS_ * INNER_ * sizeof(half_t));
  half_t* VT    = (half_t*)alloc((size_t)ROWS_ * INNER_ * sizeof(half_t));
  half_t* attnH = (half_t*)alloc((size_t)ROWS_ * INNER_ * sizeof(half_t));

  cvt_f16_kernel<<<2048, 256, 0, stream>>>(x, xh, ROWS_ * D_);
  transpose_f16_kernel<<<2048, 256, 0, stream>>>(Wq, wqT, D_, INNER_);
  transpose_f16_kernel<<<2048, 256, 0, stream>>>(Wk, wkT, D_, INNER_);
  transpose_f16_kernel<<<2048, 256, 0, stream>>>(Wv, wvT, D_, INNER_);
  transpose_f16_kernel<<<2048, 256, 0, stream>>>(Wo, woT, INNER_, D_);

  // (ROWS/32)*(INNER/16) = 16384 wave-tiles, 4 waves per block
  qkv_kernel<<<(ROWS_ / 32) * (INNER_ / 16) / 4, 128, 0, stream>>>(
      xh, wqT, wkT, wvT, Q, K, VT);

  // B*H*(T/64) blocks, each = 4 waves sharing K/V chunks via async LDS staging
  attn_kernel<<<B_ * H_ * (T_ / 64), 128, 0, stream>>>(Q, K, VT, mask, attnH);

  // (ROWS/32)*(D/32) = 8192 wave-tiles
  out_proj_kernel<<<(ROWS_ / 32) * (D_ / 32) / 4, 128, 0, stream>>>(
      attnH, woT, bo, out);
}
